// Model_10033043603480
// MI455X (gfx1250) — matrix operand order repaired and checked by reading, not compiled, not run on hardware
//
#include <hip/hip_runtime.h>
#include <hip/hip_bf16.h>
#include <stdint.h>

typedef __attribute__((ext_vector_type(16))) _Float16 v16h;
typedef __attribute__((ext_vector_type(8)))  float    v8f;

#define WAVES 8               // 8 wave32s = 256 threads per block
#define GRID_CAP 512          // persistent blocks; weights staged once per block
static constexpr float EPSV = 0.1f;

// ---------------- wave-local LDS fence (store<->load within one wave) ------
__device__ __forceinline__ void wave_lds_fence() {
    __builtin_amdgcn_wave_barrier();
    asm volatile("s_wait_dscnt 0" ::: "memory");
    __builtin_amdgcn_wave_barrier();
}

// ---------------- fragment builders (CDNA5 16x16x32 f16 WMMA layouts) ------
// A 16x32 (MxK): lane holds row M=lane%16; halves 0-7 = K klo..klo+7,
// halves 8-15 = K 16+klo..; klo = (lane>=16)?8:0.
__device__ __forceinline__ v16h a_frag_f32(const float* __restrict__ p) {
    const int lane = threadIdx.x & 31;
    const int klo = (lane & 16) ? 8 : 0;
    v16h a;
#pragma unroll
    for (int i = 0; i < 8; ++i) a[i] = (_Float16)p[klo + i];
#pragma unroll
    for (int i = 0; i < 8; ++i) a[8 + i] = (_Float16)p[16 + klo + i];
    return a;
}
__device__ __forceinline__ v16h a_frag_f16(const _Float16* __restrict__ p) {
    const int lane = threadIdx.x & 31;
    const int klo = (lane & 16) ? 8 : 0;
    v16h a;
#pragma unroll
    for (int i = 0; i < 8; ++i) a[i] = p[klo + i];
#pragma unroll
    for (int i = 0; i < 8; ++i) a[8 + i] = p[16 + klo + i];
    return a;
}
// B 32x16 (KxN) from LDS weight stored transposed [n][K]: lane holds
// column N = ngrp*16 + lane%16, K = kchunk*32 + klo + 0..7 and kchunk*32 + 16 + klo + 0..7; klo = (lane>=16)?8:0.
__device__ __forceinline__ v16h b_frag(const _Float16* __restrict__ wT, int K,
                                       int ngrp, int kchunk) {
    const int lane = threadIdx.x & 31;
    const _Float16* p = wT + (ngrp * 16 + (lane & 15)) * K
                           + kchunk * 32 + ((lane & 16) ? 8 : 0);
    v16h b;
#pragma unroll
    for (int i = 0; i < 8; ++i) { b[i] = p[i]; b[8 + i] = p[16 + i]; }
    return b;
}

#define WMMA(a, b, c) \
    __builtin_amdgcn_wmma_f32_16x16x32_f16(false, (a), false, (b), (short)0, (c), false, false)

// C-layout epilogue: +bias, ReLU, write f16 tile [16][64] (row-major) to LDS.
__device__ __forceinline__ void epi_relu_to_tile(const v8f acc[4],
                                                 const float* __restrict__ bias,
                                                 _Float16* __restrict__ tp) {
    const int lane = threadIdx.x & 31;
    const int mhi = (lane & 16) ? 8 : 0;
    const int nlo = lane & 15;
#pragma unroll
    for (int g = 0; g < 4; ++g)
#pragma unroll
        for (int r = 0; r < 8; ++r) {
            float v = acc[g][r] + bias[g * 16 + nlo];
            tp[(r + mhi) * 64 + g * 16 + nlo] = (_Float16)fmaxf(v, 0.0f);
        }
}

// Preload a 64x64 weight (LDS-transposed) into 8 register B-fragments.
__device__ __forceinline__ void preload_b64(const _Float16* __restrict__ wT, v16h bf[8]) {
#pragma unroll
    for (int c = 0; c < 2; ++c)
#pragma unroll
        for (int g = 0; g < 4; ++g) bf[c * 4 + g] = b_frag(wT, 64, g, c);
}

// Second GEMM: T(16x64 f16 in LDS) x W2 (register fragments) -> o[4]
__device__ __forceinline__ void gemm2_reg(const _Float16* __restrict__ tp,
                                          const v16h bf[8], v8f o[4]) {
    const int lane = threadIdx.x & 31;
    const _Float16* trow = tp + (lane & 15) * 64;
#pragma unroll
    for (int c = 0; c < 2; ++c) {
        v16h a = a_frag_f16(trow + c * 32);
#pragma unroll
        for (int g = 0; g < 4; ++g) o[g] = WMMA(a, bf[c * 4 + g], o[g]);
    }
}

// ------------------------------------------------------------------ kernels
// node init: out = relu(X@W1+b1)@W2+b2, X = emb[n_id], H=64
__global__ __launch_bounds__(32 * WAVES)
void node_init_kernel(const float* __restrict__ emb, const int* __restrict__ n_id,
                      const float* __restrict__ W1, const float* __restrict__ b1,
                      const float* __restrict__ W2, const float* __restrict__ b2,
                      float* __restrict__ out, int ntiles) {
    __shared__ _Float16 sW1[64 * 64];
    __shared__ _Float16 sW2[64 * 64];
    __shared__ float sb1[64], sb2[64];
    __shared__ _Float16 tile[WAVES][16 * 64];
    for (int idx = threadIdx.x; idx < 64 * 64; idx += blockDim.x) {
        int k = idx >> 6, n = idx & 63;
        sW1[n * 64 + k] = (_Float16)W1[idx];
        sW2[n * 64 + k] = (_Float16)W2[idx];
    }
    if (threadIdx.x < 64) { sb1[threadIdx.x] = b1[threadIdx.x]; sb2[threadIdx.x] = b2[threadIdx.x]; }
    __syncthreads();
    const int wid = threadIdx.x >> 5;
    const int lane = threadIdx.x & 31;
    const int mhi = (lane & 16) ? 8 : 0;
    const int nlo = lane & 15;
    v16h bf1[8], bf2[8];
    preload_b64(sW1, bf1);          // stage-1 weights register-resident
    preload_b64(sW2, bf2);          // stage-2 weights register-resident
    _Float16* tp = tile[wid];
    for (int t = blockIdx.x * WAVES + wid; t < ntiles; t += gridDim.x * WAVES) {
        const float* xrow = emb + (size_t)n_id[t * 16 + (lane & 15)] * 64;
        v8f acc[4] = {};
#pragma unroll
        for (int c = 0; c < 2; ++c) {
            v16h a = a_frag_f32(xrow + c * 32);
#pragma unroll
            for (int g = 0; g < 4; ++g) acc[g] = WMMA(a, bf1[c * 4 + g], acc[g]);
        }
        epi_relu_to_tile(acc, sb1, tp);
        wave_lds_fence();
        v8f o[4] = {};
        gemm2_reg(tp, bf2, o);
#pragma unroll
        for (int g = 0; g < 4; ++g)
#pragma unroll
            for (int r = 0; r < 8; ++r)
                out[(size_t)(t * 16 + r + mhi) * 64 + g * 16 + nlo] = o[g][r] + sb2[g * 16 + nlo];
        wave_lds_fence();           // WAR: tile rewritten next iteration
    }
}

// fused edge init: reads edge_attr once, produces eu16 and eb16 (f16 rows).
// Dynamic LDS: sW1u[64*256], sW1b[64*256], sW2u[64*64], sW2b[64*64],
// tile[WAVES][16*64], 4x64 f32 biases.
__global__ __launch_bounds__(32 * WAVES)
void edge_init_kernel(const float* __restrict__ eattr,
                      const float* __restrict__ uW1, const float* __restrict__ ub1,
                      const float* __restrict__ uW2, const float* __restrict__ ub2,
                      const float* __restrict__ bW1, const float* __restrict__ bb1,
                      const float* __restrict__ bW2, const float* __restrict__ bb2,
                      _Float16* __restrict__ outu, _Float16* __restrict__ outb, int ntiles) {
    extern __shared__ char smem[];
    _Float16* sW1u = (_Float16*)smem;
    _Float16* sW1b = sW1u + 64 * 256;
    _Float16* sW2u = sW1b + 64 * 256;
    _Float16* sW2b = sW2u + 64 * 64;
    _Float16* tileA = sW2b + 64 * 64;            // WAVES*16*64
    float* sb1u = (float*)(tileA + WAVES * 16 * 64);
    float* sb2u = sb1u + 64;
    float* sb1b = sb2u + 64;
    float* sb2b = sb1b + 64;
    for (int idx = threadIdx.x; idx < 256 * 64; idx += blockDim.x) {
        int k = idx >> 6, n = idx & 63;
        sW1u[n * 256 + k] = (_Float16)uW1[idx];
        sW1b[n * 256 + k] = (_Float16)bW1[idx];
    }
    for (int idx = threadIdx.x; idx < 64 * 64; idx += blockDim.x) {
        int k = idx >> 6, n = idx & 63;
        sW2u[n * 64 + k] = (_Float16)uW2[idx];
        sW2b[n * 64 + k] = (_Float16)bW2[idx];
    }
    if (threadIdx.x < 64) {
        sb1u[threadIdx.x] = ub1[threadIdx.x]; sb2u[threadIdx.x] = ub2[threadIdx.x];
        sb1b[threadIdx.x] = bb1[threadIdx.x]; sb2b[threadIdx.x] = bb2[threadIdx.x];
    }
    __syncthreads();
    const int wid = threadIdx.x >> 5;
    const int lane = threadIdx.x & 31;
    const int mhi = (lane & 16) ? 8 : 0;
    const int nlo = lane & 15;
    v16h bfu2[8], bfb2[8];
    preload_b64(sW2u, bfu2);
    preload_b64(sW2b, bfb2);
    _Float16* tp = tileA + wid * 16 * 64;
    for (int t = blockIdx.x * WAVES + wid; t < ntiles; t += gridDim.x * WAVES) {
        const float* xrow = eattr + (size_t)(t * 16 + (lane & 15)) * 256;
        v8f au[4] = {}, ab[4] = {};
#pragma unroll
        for (int c = 0; c < 8; ++c) {
            v16h a = a_frag_f32(xrow + c * 32);
#pragma unroll
            for (int g = 0; g < 4; ++g) {
                au[g] = WMMA(a, b_frag(sW1u, 256, g, c), au[g]);
                ab[g] = WMMA(a, b_frag(sW1b, 256, g, c), ab[g]);
            }
        }
        // user->book edge features
        epi_relu_to_tile(au, sb1u, tp);
        wave_lds_fence();
        {
            v8f o[4] = {};
            gemm2_reg(tp, bfu2, o);
#pragma unroll
            for (int g = 0; g < 4; ++g)
#pragma unroll
                for (int r = 0; r < 8; ++r)
                    outu[(size_t)(t * 16 + r + mhi) * 64 + g * 16 + nlo] =
                        (_Float16)(o[g][r] + sb2u[g * 16 + nlo]);
        }
        wave_lds_fence();
        // book->user edge features
        epi_relu_to_tile(ab, sb1b, tp);
        wave_lds_fence();
        {
            v8f o[4] = {};
            gemm2_reg(tp, bfb2, o);
#pragma unroll
            for (int g = 0; g < 4; ++g)
#pragma unroll
                for (int r = 0; r < 8; ++r)
                    outb[(size_t)(t * 16 + r + mhi) * 64 + g * 16 + nlo] =
                        (_Float16)(o[g][r] + sb2b[g * 16 + nlo]);
        }
        wave_lds_fence();
    }
}

// conv message + scatter: msg = relu(cat[hf[idxf], hs[idxs], e]@W1+b1)@W2+b2;
// atomicAdd into acc[idxf].
__global__ __launch_bounds__(32 * WAVES)
void conv_scatter_kernel(const float* __restrict__ hfirst, const int* __restrict__ idxfirst,
                         const float* __restrict__ hsecond, const int* __restrict__ idxsecond,
                         const _Float16* __restrict__ efeat,
                         const float* __restrict__ W1, const float* __restrict__ b1,
                         const float* __restrict__ W2, const float* __restrict__ b2,
                         float* __restrict__ acc_out, int ntiles) {
    __shared__ _Float16 sW1[64 * 192];
    __shared__ _Float16 sW2[64 * 64];
    __shared__ float sb1[64], sb2[64];
    __shared__ _Float16 tile[WAVES][16 * 64];
    for (int idx = threadIdx.x; idx < 192 * 64; idx += blockDim.x) {
        int k = idx >> 6, n = idx & 63;
        sW1[n * 192 + k] = (_Float16)W1[idx];
    }
    for (int idx = threadIdx.x; idx < 64 * 64; idx += blockDim.x) {
        int k = idx >> 6, n = idx & 63;
        sW2[n * 64 + k] = (_Float16)W2[idx];
    }
    if (threadIdx.x < 64) { sb1[threadIdx.x] = b1[threadIdx.x]; sb2[threadIdx.x] = b2[threadIdx.x]; }
    __syncthreads();
    const int wid = threadIdx.x >> 5;
    const int lane = threadIdx.x & 31;
    const int mhi = (lane & 16) ? 8 : 0;
    const int nlo = lane & 15;
    v16h bf2[8];
    preload_b64(sW2, bf2);
    _Float16* tp = tile[wid];
    for (int t = blockIdx.x * WAVES + wid; t < ntiles; t += gridDim.x * WAVES) {
        const int e = t * 16 + (lane & 15);
        const float*    xA = hfirst  + (size_t)idxfirst[e]  * 64;
        const float*    xB = hsecond + (size_t)idxsecond[e] * 64;
        const _Float16* xE = efeat   + (size_t)e * 64;
        v8f acc[4] = {};
#pragma unroll
        for (int c = 0; c < 2; ++c) {
            v16h a = a_frag_f32(xA + c * 32);
#pragma unroll
            for (int g = 0; g < 4; ++g) acc[g] = WMMA(a, b_frag(sW1, 192, g, c), acc[g]);
        }
#pragma unroll
        for (int c = 0; c < 2; ++c) {
            v16h a = a_frag_f32(xB + c * 32);
#pragma unroll
            for (int g = 0; g < 4; ++g) acc[g] = WMMA(a, b_frag(sW1, 192, g, 2 + c), acc[g]);
        }
#pragma unroll
        for (int c = 0; c < 2; ++c) {
            v16h a = a_frag_f16(xE + c * 32);
#pragma unroll
            for (int g = 0; g < 4; ++g) acc[g] = WMMA(a, b_frag(sW1, 192, g, 4 + c), acc[g]);
        }
        epi_relu_to_tile(acc, sb1, tp);
        wave_lds_fence();
        v8f o[4] = {};
        gemm2_reg(tp, bf2, o);
#pragma unroll
        for (int r = 0; r < 8; ++r) {
            const int m = r + mhi;
            float* basep = acc_out + (size_t)idxfirst[t * 16 + m] * 64;
#pragma unroll
            for (int g = 0; g < 4; ++g) {
                float v = o[g][r] + sb2[g * 16 + nlo];
                __hip_atomic_fetch_add(basep + g * 16 + nlo, v,
                                       __ATOMIC_RELAXED, __HIP_MEMORY_SCOPE_AGENT);
            }
        }
        wave_lds_fence();
    }
}

// link classifier: out = relu(cat[hu[i0], hb[i1]]@W1+b1) . w2 + b2
__global__ __launch_bounds__(32 * WAVES)
void classifier_kernel(const float* __restrict__ hu, const float* __restrict__ hb,
                       const int* __restrict__ eli, int ELn,
                       const float* __restrict__ W1, const float* __restrict__ b1,
                       const float* __restrict__ w2, const float* __restrict__ b2,
                       float* __restrict__ out, int ntiles) {
    __shared__ _Float16 sW1[64 * 128];
    __shared__ float sb1[64];
    __shared__ float sw2[64];
    __shared__ float tile[WAVES][16 * 64];
    for (int idx = threadIdx.x; idx < 128 * 64; idx += blockDim.x) {
        int k = idx >> 6, n = idx & 63;
        sW1[n * 128 + k] = (_Float16)W1[idx];
    }
    if (threadIdx.x < 64) { sb1[threadIdx.x] = b1[threadIdx.x]; sw2[threadIdx.x] = w2[threadIdx.x]; }
    __syncthreads();
    const int wid = threadIdx.x >> 5;
    const int lane = threadIdx.x & 31;
    const int mhi = (lane & 16) ? 8 : 0;
    const int nlo = lane & 15;
    for (int t = blockIdx.x * WAVES + wid; t < ntiles; t += gridDim.x * WAVES) {
        const int e = t * 16 + (lane & 15);
        const float* xU = hu + (size_t)eli[e] * 64;
        const float* xB = hb + (size_t)eli[ELn + e] * 64;
        v8f acc[4] = {};
#pragma unroll
        for (int c = 0; c < 2; ++c) {
            v16h a = a_frag_f32(xU + c * 32);
#pragma unroll
            for (int g = 0; g < 4; ++g) acc[g] = WMMA(a, b_frag(sW1, 128, g, c), acc[g]);
        }
#pragma unroll
        for (int c = 0; c < 2; ++c) {
            v16h a = a_frag_f32(xB + c * 32);
#pragma unroll
            for (int g = 0; g < 4; ++g) acc[g] = WMMA(a, b_frag(sW1, 128, g, 2 + c), acc[g]);
        }
#pragma unroll
        for (int g = 0; g < 4; ++g)
#pragma unroll
            for (int r = 0; r < 8; ++r)
                tile[wid][(r + mhi) * 64 + g * 16 + nlo] =
                    fmaxf(acc[g][r] + sb1[g * 16 + nlo], 0.0f);
        wave_lds_fence();
        if (lane < 16) {
            const float* tr = tile[wid] + lane * 64;
            float s = b2[0];
#pragma unroll
            for (int n = 0; n < 64; ++n) s += tr[n] * sw2[n];
            out[t * 16 + lane] = s;
        }
        wave_lds_fence();
    }
}

__global__ void scale_kernel(float* __restrict__ dst, const float* __restrict__ src,
                             float f, int n) {
    int i = blockIdx.x * blockDim.x + threadIdx.x;
    if (i < n) dst[i] = f * src[i];
}
__global__ void relu_copy_kernel(float* __restrict__ dst, const float* __restrict__ src, int n) {
    int i = blockIdx.x * blockDim.x + threadIdx.x;
    if (i < n) dst[i] = fmaxf(src[i], 0.0f);
}

// ------------------------------------------------------------------ launch
static inline int grid_for(int tiles) {
    int b = (tiles + WAVES - 1) / WAVES;
    return b < GRID_CAP ? b : GRID_CAP;
}

extern "C" void kernel_launch(void* const* d_in, const int* in_sizes, int n_in,
                              void* d_out, int out_size, void* d_ws, size_t ws_size,
                              hipStream_t stream) {
    (void)in_sizes; (void)n_in; (void)out_size; (void)ws_size;
    const float* emb_user = (const float*)d_in[0];
    const float* emb_book = (const float*)d_in[1];
    const float* nu_W1 = (const float*)d_in[2];  const float* nu_b1 = (const float*)d_in[3];
    const float* nu_W2 = (const float*)d_in[4];  const float* nu_b2 = (const float*)d_in[5];
    const float* nb_W1 = (const float*)d_in[6];  const float* nb_b1 = (const float*)d_in[7];
    const float* nb_W2 = (const float*)d_in[8];  const float* nb_b2 = (const float*)d_in[9];
    const float* eu_W1 = (const float*)d_in[10]; const float* eu_b1 = (const float*)d_in[11];
    const float* eu_W2 = (const float*)d_in[12]; const float* eu_b2 = (const float*)d_in[13];
    const float* eb_W1 = (const float*)d_in[14]; const float* eb_b1 = (const float*)d_in[15];
    const float* eb_W2 = (const float*)d_in[16]; const float* eb_b2 = (const float*)d_in[17];
    const float* cu2b_W1 = (const float*)d_in[18]; const float* cu2b_b1 = (const float*)d_in[19];
    const float* cu2b_W2 = (const float*)d_in[20]; const float* cu2b_b2 = (const float*)d_in[21];
    const float* cb2u_W1 = (const float*)d_in[22]; const float* cb2u_b1 = (const float*)d_in[23];
    const float* cb2u_W2 = (const float*)d_in[24]; const float* cb2u_b2 = (const float*)d_in[25];
    const float* cls_W1 = (const float*)d_in[26]; const float* cls_b1 = (const float*)d_in[27];
    const float* cls_W2 = (const float*)d_in[28]; const float* cls_b2 = (const float*)d_in[29];
    const float* edge_attr = (const float*)d_in[30];
    const int* n_id_user = (const int*)d_in[31];
    const int* n_id_book = (const int*)d_in[32];
    const int* edge_index = (const int*)d_in[33];
    const int* eli = (const int*)d_in[34];

    const int NUn = 100000, NBn = 100000, En = 300000, ELn = 150000;
    const int* esrc = edge_index;       // user indices
    const int* edst = edge_index + En;  // book indices

    char* w = (char*)d_ws;
    float* hu   = (float*)w; w += (size_t)NUn * 64 * sizeof(float);
    float* hb   = (float*)w; w += (size_t)NBn * 64 * sizeof(float);
    float* accu = (float*)w; w += (size_t)NUn * 64 * sizeof(float);
    float* accb = (float*)w; w += (size_t)NBn * 64 * sizeof(float);
    _Float16* eu16 = (_Float16*)w; w += (size_t)En * 64 * sizeof(_Float16);
    _Float16* eb16 = (_Float16*)w; w += (size_t)En * 64 * sizeof(_Float16);

    dim3 blk(32 * WAVES);
    const int tilesU = NUn / 16, tilesB = NBn / 16, tilesE = En / 16, tilesL = ELn / 16;

    node_init_kernel<<<grid_for(tilesU), blk, 0, stream>>>(
        emb_user, n_id_user, nu_W1, nu_b1, nu_W2, nu_b2, hu, tilesU);
    node_init_kernel<<<grid_for(tilesB), blk, 0, stream>>>(
        emb_book, n_id_book, nb_W1, nb_b1, nb_W2, nb_b2, hb, tilesB);

    size_t einit_lds = (size_t)(2 * 64 * 256 + 2 * 64 * 64 + WAVES * 16 * 64) * sizeof(_Float16)
                     + 4 * 64 * sizeof(float);
    edge_init_kernel<<<grid_for(tilesE), blk, einit_lds, stream>>>(
        edge_attr, eu_W1, eu_b1, eu_W2, eu_b2, eb_W1, eb_b1, eb_W2, eb_b2, eu16, eb16, tilesE);

    const int nU = NUn * 64, nB = NBn * 64;
    for (int l = 0; l < 2; ++l) {
        const float fac = (l == 0) ? (2.0f + EPSV) : (1.0f + EPSV);
        scale_kernel<<<(nB + 255) / 256, 256, 0, stream>>>(accb, hb, fac, nB);
        scale_kernel<<<(nU + 255) / 256, 256, 0, stream>>>(accu, hu, fac, nU);
        conv_scatter_kernel<<<grid_for(tilesE), blk, 0, stream>>>(
            hb, edst, hu, esrc, eu16,
            cu2b_W1 + (size_t)l * 192 * 64, cu2b_b1 + l * 64,
            cu2b_W2 + (size_t)l * 64 * 64,  cu2b_b2 + l * 64, accb, tilesE);
        conv_scatter_kernel<<<grid_for(tilesE), blk, 0, stream>>>(
            hu, esrc, hb, edst, eb16,
            cb2u_W1 + (size_t)l * 192 * 64, cb2u_b1 + l * 64,
            cb2u_W2 + (size_t)l * 64 * 64,  cb2u_b2 + l * 64, accu, tilesE);
        if (l == 0) {
            relu_copy_kernel<<<(nB + 255) / 256, 256, 0, stream>>>(hb, accb, nB);
            relu_copy_kernel<<<(nU + 255) / 256, 256, 0, stream>>>(hu, accu, nU);
        }
    }
    classifier_kernel<<<grid_for(tilesL), blk, 0, stream>>>(
        accu, accb, eli, ELn, cls_W1, cls_b1, cls_W2, cls_b2, (float*)d_out, tilesL);
}
